// RecurrentAttentionNetwork_13975823581926
// MI455X (gfx1250) — compile-verified, matrix operand order checked by reading, not run on hardware
//
#include <hip/hip_runtime.h>
#include <hip/hip_bf16.h>

typedef __attribute__((ext_vector_type(16))) __bf16 v16bf;
typedef __attribute__((ext_vector_type(8)))  float  v8f;
typedef __attribute__((ext_vector_type(4)))  unsigned int v4u;
typedef __attribute__((ext_vector_type(8)))  int v8i;
typedef __attribute__((ext_vector_type(4)))  int v4i;

#if defined(__has_builtin)
#if __has_builtin(__builtin_amdgcn_tensor_load_to_lds) && __has_builtin(__builtin_amdgcn_s_wait_tensorcnt)
#define HAVE_TDM 1
#endif
#endif
#ifndef HAVE_TDM
#define HAVE_TDM 0
#endif

constexpr int NB   = 32;     // batch
constexpr int NT   = 1024;   // sequence length
constexpr int NTT  = 4;      // target width
constexpr int NE   = 300;    // embedding dim
constexpr int NEP  = 320;    // padded embedding dim (multiple of 32 for WMMA-K)
constexpr int NH   = 128;    // LSTM hidden
constexpr int N4H  = 512;    // 4*H gate width
constexpr int NG   = 128;    // episode dim
constexpr int NDM  = 257;    // memory feature dim (2H+1)
constexpr int DIN2 = 256;    // layer-1 input dim

__device__ __forceinline__ float sigmoidf(float x){ return 1.0f/(1.0f+__expf(-x)); }

// K index inside a 32-wide chunk for 16-bit A/B WMMA fragments (ISA 7.12.2):
// vgpr 0..3 -> K 0..7 (+8*lanehalf), vgpr 4..7 -> K 16..23 (+8*lanehalf)
__device__ __forceinline__ int wmma_k(int e, int kh){
  int v = e >> 1, h = e & 1;
  return ((v >> 2) << 4) + (kh << 3) + ((v & 3) << 1) + h;
}

// ---------------- weight / activation casting ----------------

__global__ void k_cast_w(const float* __restrict__ src, __bf16* __restrict__ dst,
                         int K, int Kp, int N){
  int idx = blockIdx.x * blockDim.x + threadIdx.x;
  if (idx >= Kp * N) return;
  int k = idx / N, n = idx % N;
  float v = (k < K) ? src[(size_t)k * N + n] : 0.0f;
  dst[idx] = (__bf16)v;
}

__global__ void k_embed_sent(const int* __restrict__ ids, const float* __restrict__ emb,
                             __bf16* __restrict__ out){
  int bt = blockIdx.x;          // 0..NB*NT-1
  int e  = threadIdx.x;         // 320 threads
  int row = ids[bt];
  float v = (e < NE) ? emb[(size_t)row * NE + e] : 0.0f;
  out[(size_t)bt * NEP + e] = (__bf16)v;
}

__global__ void k_vtarget(const int* __restrict__ tids, const float* __restrict__ emb,
                          float* __restrict__ vt){
  int b = blockIdx.x, e = threadIdx.x;   // 320 threads
  if (e >= NE) return;
  float s = 0.0f;
  for (int i = 0; i < NTT; ++i) s += emb[(size_t)tids[b*NTT+i] * NE + e];
  vt[(size_t)b * NE + e] = s * (1.0f / NTT);
}

// ---------------- bf16 WMMA GEMM: C[M,512] = A[M,Kp] * W[Kp,512] + bias -----
// block = 256 threads = 8 waves; block tile 128x128; wave tile 16x128.
// A tile (128x32 bf16 = 8KB) is DMA'd into LDS by the Tensor Data Mover;
// B tile is staged transposed by the waves (TDM cannot transpose).

__global__ __launch_bounds__(256) void k_gemm_xw(
    const __bf16* __restrict__ A, const __bf16* __restrict__ W,
    const float* __restrict__ bias, float* __restrict__ C, int Kp){
  __shared__ __bf16 sA[128 * 32];   // [row][k]
  __shared__ __bf16 sBt[128 * 32];  // [n][k] (transposed)
  const int tid  = threadIdx.x;
  const int lane = tid & 31;
  const int wv   = tid >> 5;
  const int m0   = blockIdx.y * 128;
  const int n0   = blockIdx.x * 128;
  const int lm   = lane & 15;
  const int kh   = lane >> 4;

  v8f acc[8] = {};

  for (int kk = 0; kk < Kp; kk += 32) {
#if HAVE_TDM
    // ---- Tensor Data Mover: 2D tile (32 x 128 rows) global -> LDS ----
    if (wv == 0) {
      unsigned long long ga =
          (unsigned long long)(uintptr_t)(A + (size_t)m0 * Kp + kk);
      unsigned lds = (unsigned)(uintptr_t)&sA[0];   // low 32 bits = LDS offset
      v4u g0;
      g0[0] = 1u;                                        // count=1, user mode
      g0[1] = lds;                                       // lds_addr
      g0[2] = (unsigned)(ga & 0xffffffffu);              // global_addr[31:0]
      g0[3] = (unsigned)((ga >> 32) & 0x01ffffffu)       // global_addr[56:32]
              | 0x80000000u;                             // type=2 ("image")
      v8i g1;
      g1[0] = (int)(1u << 16);                           // data_size=1 -> 2 bytes
      g1[1] = (int)(((unsigned)Kp & 0xffffu) << 16);     // tensor_dim0 lo
      g1[2] = (int)(((unsigned)Kp >> 16)                 // tensor_dim0 hi
              | (((unsigned)(NB * NT) & 0xffffu) << 16));// tensor_dim1 lo
      g1[3] = (int)(((unsigned)(NB * NT) >> 16)          // tensor_dim1 hi
              | (32u << 16));                            // tile_dim0 = 32
      g1[4] = 128;                                       // tile_dim1 = 128
      g1[5] = Kp;                                        // tensor_dim0_stride lo
      g1[6] = 0;                                         // stride hi / dim1_stride lo
      g1[7] = 0;
      v4i z4 = {0, 0, 0, 0};
#if __clang_major__ >= 23
      v8i z8 = {0, 0, 0, 0, 0, 0, 0, 0};
      __builtin_amdgcn_tensor_load_to_lds(g0, g1, z4, z4, z8, 0);
#else
      __builtin_amdgcn_tensor_load_to_lds(g0, g1, z4, z4, 0);
#endif
    }
#else
    #pragma unroll
    for (int i = 0; i < 16; ++i) {
      int idx = i * 256 + tid;
      int r = idx >> 5, k = idx & 31;
      sA[idx] = A[(size_t)(m0 + r) * Kp + kk + k];
    }
#endif
    #pragma unroll
    for (int i = 0; i < 16; ++i) {
      int idx = i * 256 + tid;
      int n = idx & 127, k = idx >> 7;
      sBt[n * 32 + k] = W[(size_t)(kk + k) * N4H + n0 + n];
    }
    if (kk + 32 < Kp)  // hint next A tile toward L2 (global_prefetch_b8)
      __builtin_prefetch(&A[(size_t)(m0 + (tid >> 1)) * Kp + kk + 32], 0, 1);
#if HAVE_TDM
    if (wv == 0) __builtin_amdgcn_s_wait_tensorcnt(0);
#endif
    __syncthreads();

    v16bf af;
    #pragma unroll
    for (int e = 0; e < 16; ++e)
      af[e] = sA[(wv * 16 + lm) * 32 + wmma_k(e, kh)];

    #pragma unroll
    for (int nt = 0; nt < 8; ++nt) {
      v16bf bfv;
      #pragma unroll
      for (int e = 0; e < 16; ++e)
        bfv[e] = sBt[(nt * 16 + lm) * 32 + wmma_k(e, kh)];
      acc[nt] = __builtin_amdgcn_wmma_f32_16x16x32_bf16(
          false, af, false, bfv, (short)0, acc[nt], false, false);
    }
    __syncthreads();
  }

  #pragma unroll
  for (int nt = 0; nt < 8; ++nt) {
    #pragma unroll
    for (int r = 0; r < 8; ++r) {
      int row = m0 + wv * 16 + r + 8 * kh;
      int col = n0 + nt * 16 + lm;
      C[(size_t)row * N4H + col] = acc[nt][r] + bias[col];
    }
  }
}

// ---------------- persistent LSTM scan ----------------
// block 0 = forward, block 1 = backward. 256 threads = 8 waves.
// Wh bf16 fragments live in VGPRs for all 1024 steps; h (bf16), c, g in LDS.

__global__ __launch_bounds__(256) void k_lstm(
    const float* __restrict__ xw_f, const float* __restrict__ xw_b,
    const __bf16* __restrict__ whf, const __bf16* __restrict__ whb,
    float* __restrict__ hfo, float* __restrict__ hbo){
  extern __shared__ char smem[];
  float*  g   = (float*)smem;                               // 32*512 f32 = 64KB
  float*  cst = (float*)(smem + NB*N4H*4);                  // 32*128 f32 = 16KB
  __bf16* hbf = (__bf16*)(smem + NB*N4H*4 + NB*NH*4);       // 32*128 bf16 = 8KB

  const int rev = blockIdx.x;
  const float*  xw   = rev ? xw_b : xw_f;
  const __bf16* wh   = rev ? whb  : whf;
  float*        hout = rev ? hbo  : hfo;

  const int tid = threadIdx.x, lane = tid & 31, wv = tid >> 5;
  const int lm = lane & 15, kh = lane >> 4;

  for (int i = tid; i < NB*NH; i += 256) { cst[i] = 0.0f; hbf[i] = (__bf16)0.0f; }
  __syncthreads();

  // Preload this wave's Wh fragments (4 n-tiles x 4 k-steps) into registers.
  v16bf bfrag[4][4];
  #pragma unroll
  for (int ntl = 0; ntl < 4; ++ntl) {
    int ntg = wv * 4 + ntl;
    #pragma unroll
    for (int ks = 0; ks < 4; ++ks) {
      #pragma unroll
      for (int e = 0; e < 16; ++e) {
        int k = ks * 32 + wmma_k(e, kh);
        bfrag[ntl][ks][e] = wh[(size_t)k * N4H + ntg * 16 + lm];
      }
    }
  }

  for (int s = 0; s < NT; ++s) {
    int t = rev ? (NT - 1 - s) : s;

    // Init accumulators with the precomputed input projection xW[:, t, :].
    v8f acc[2][4];
    #pragma unroll
    for (int mt = 0; mt < 2; ++mt)
      #pragma unroll
      for (int ntl = 0; ntl < 4; ++ntl) {
        int ntg = wv * 4 + ntl;
        #pragma unroll
        for (int r = 0; r < 8; ++r) {
          int bb = mt * 16 + r + 8 * kh;
          acc[mt][ntl][r] = xw[((size_t)bb * NT + t) * N4H + ntg * 16 + lm];
        }
      }

    // g += h @ Wh   (M=32, N=512, K=128 in 4 bf16 WMMA k-steps)
    #pragma unroll
    for (int ks = 0; ks < 4; ++ks) {
      v16bf af[2];
      #pragma unroll
      for (int mt = 0; mt < 2; ++mt)
        #pragma unroll
        for (int e = 0; e < 16; ++e) {
          int k = ks * 32 + wmma_k(e, kh);
          af[mt][e] = hbf[(mt * 16 + lm) * NH + k];
        }
      #pragma unroll
      for (int ntl = 0; ntl < 4; ++ntl)
        #pragma unroll
        for (int mt = 0; mt < 2; ++mt)
          acc[mt][ntl] = __builtin_amdgcn_wmma_f32_16x16x32_bf16(
              false, af[mt], false, bfrag[ntl][ks], (short)0, acc[mt][ntl],
              false, false);
    }

    // spill gates to LDS for cross-tile mixing
    #pragma unroll
    for (int mt = 0; mt < 2; ++mt)
      #pragma unroll
      for (int ntl = 0; ntl < 4; ++ntl) {
        int ntg = wv * 4 + ntl;
        #pragma unroll
        for (int r = 0; r < 8; ++r) {
          int bb = mt * 16 + r + 8 * kh;
          g[bb * N4H + ntg * 16 + lm] = acc[mt][ntl][r];
        }
      }
    __syncthreads();

    for (int i = tid; i < NB*NH; i += 256) {
      int bb = i >> 7, hc = i & 127;
      float gi = g[bb*N4H + hc];
      float gj = g[bb*N4H + NH + hc];
      float gf = g[bb*N4H + 2*NH + hc];
      float go = g[bb*N4H + 3*NH + hc];
      float cc = sigmoidf(gf + 1.0f) * cst[i] + sigmoidf(gi) * tanhf(gj);
      float hh = sigmoidf(go) * tanhf(cc);
      cst[i] = cc;
      hbf[i] = (__bf16)hh;
      hout[((size_t)bb * NT + t) * NH + hc] = hh;
    }
    __syncthreads();
  }
}

// ---------------- layer glue / memory construction ----------------

__global__ void k_concat_x1(const float* __restrict__ hf, const float* __restrict__ hb,
                            __bf16* __restrict__ x1){
  int bt = blockIdx.x, d = threadIdx.x;   // 256 threads
  float v = (d < NH) ? hf[(size_t)bt*NH + d] : hb[(size_t)bt*NH + d - NH];
  x1[(size_t)bt * DIN2 + d] = (__bf16)v;
}

__global__ void k_memory(const float* __restrict__ hf, const float* __restrict__ hb,
                         const int* __restrict__ toff, float* __restrict__ mem){
  int bt = blockIdx.x;
  int b = bt >> 10, t = bt & (NT - 1);
  int d = threadIdx.x;                     // 256 threads
  int left = toff[b], right = left + NTT;
  int dist = (t < left) ? (left - t) : ((t >= right) ? (t - right + 1) : 0);
  float u = (float)dist / (float)NT;
  float w = 1.0f - u;
  float v = (d < NH) ? hf[(size_t)bt*NH + d] : hb[(size_t)bt*NH + d - NH];
  mem[(size_t)bt * NDM + d] = v * w;
  if (d == 0) mem[(size_t)bt * NDM + 256] = u;
}

__global__ void k_zero(float* __restrict__ p, int n){
  int i = blockIdx.x * blockDim.x + threadIdx.x;
  if (i < n) p[i] = 0.0f;
}

// ---------------- episodic attention hops ----------------

__global__ void k_bconst(const float* __restrict__ ep, const float* __restrict__ vt,
                         const float* __restrict__ aw, const float* __restrict__ ab,
                         float* __restrict__ bc){
  __shared__ float red[128];
  int b = blockIdx.x, tid = threadIdx.x;   // 128 threads
  float s = ep[b*NG + tid] * aw[NDM + tid];
  for (int e = tid; e < NE; e += 128) s += vt[(size_t)b*NE + e] * aw[NDM + NG + e];
  red[tid] = s; __syncthreads();
  for (int o = 64; o > 0; o >>= 1) { if (tid < o) red[tid] += red[tid+o]; __syncthreads(); }
  if (tid == 0) bc[b] = red[0] + ab[0];
}

__global__ void k_scores(const float* __restrict__ mem, const float* __restrict__ aw,
                         const float* __restrict__ bc, float* __restrict__ sc){
  int b = blockIdx.y;
  int t = blockIdx.x * 256 + threadIdx.x;
  const float* m = mem + ((size_t)b * NT + t) * NDM;
  float s = bc[b];
  for (int d = 0; d < NDM; ++d) s += m[d] * aw[d];
  sc[b * NT + t] = tanhf(s);
}

__global__ __launch_bounds__(256) void k_softmax_content(
    const float* __restrict__ mem, const float* __restrict__ sc,
    float* __restrict__ content){
  __shared__ float p[NT];
  __shared__ float red[256];
  int b = blockIdx.x, tid = threadIdx.x;
  float mx = -1e30f;
  for (int t = tid; t < NT; t += 256) { float v = sc[b*NT + t]; p[t] = v; mx = fmaxf(mx, v); }
  red[tid] = mx; __syncthreads();
  for (int o = 128; o > 0; o >>= 1) { if (tid < o) red[tid] = fmaxf(red[tid], red[tid+o]); __syncthreads(); }
  mx = red[0]; __syncthreads();
  float sm = 0.0f;
  for (int t = tid; t < NT; t += 256) { float e = __expf(p[t] - mx); p[t] = e; sm += e; }
  red[tid] = sm; __syncthreads();
  for (int o = 128; o > 0; o >>= 1) { if (tid < o) red[tid] += red[tid+o]; __syncthreads(); }
  float inv = 1.0f / red[0];
  for (int d = tid; d < NDM; d += 256) {
    float s = 0.0f;
    for (int t = 0; t < NT; ++t) s += p[t] * mem[((size_t)b*NT + t) * NDM + d];
    content[b*NDM + d] = s * inv;
  }
}

__global__ __launch_bounds__(256) void k_hop(
    const float* __restrict__ content, const float* __restrict__ ggk,
    const float* __restrict__ ggb, const float* __restrict__ gck,
    const float* __restrict__ gcb, float* __restrict__ ep){
  __shared__ float in1[385];
  __shared__ float ru[256];
  __shared__ float in2[385];
  int b = blockIdx.x, tid = threadIdx.x;
  for (int i = tid; i < NDM; i += 256) in1[i] = content[b*NDM + i];
  if (tid < NG) in1[NDM + tid] = ep[b*NG + tid];
  __syncthreads();
  {
    float s = ggb[tid];
    for (int k = 0; k < 385; ++k) s += in1[k] * ggk[(size_t)k*256 + tid];
    ru[tid] = sigmoidf(s);
  }
  __syncthreads();
  for (int i = tid; i < NDM; i += 256) in2[i] = in1[i];
  if (tid < NG) in2[NDM + tid] = ru[tid] * in1[NDM + tid];   // r = ru[0:128]
  __syncthreads();
  if (tid < NG) {
    float s = gcb[tid];
    for (int k = 0; k < 385; ++k) s += in2[k] * gck[(size_t)k*128 + tid];
    float c = tanhf(s);
    float u = ru[NG + tid];                                   // u = ru[128:256]
    ep[b*NG + tid] = u * in1[NDM + tid] + (1.0f - u) * c;
  }
}

__global__ void k_logits(const float* __restrict__ ep, const float* __restrict__ ow,
                         const float* __restrict__ ob, float* __restrict__ out){
  int idx = blockIdx.x * blockDim.x + threadIdx.x;
  if (idx >= NB * 3) return;
  int b = idx / 3, c = idx % 3;
  float s = ob[c];
  for (int k = 0; k < NG; ++k) s += ep[b*NG + k] * ow[k*3 + c];
  out[idx] = s;
}

// ---------------- host launcher ----------------

extern "C" void kernel_launch(void* const* d_in, const int* in_sizes, int n_in,
                              void* d_out, int out_size, void* d_ws, size_t ws_size,
                              hipStream_t stream) {
  (void)in_sizes; (void)n_in; (void)out_size; (void)ws_size;
  const int*   sent_ids = (const int*)  d_in[0];
  const int*   tgt_ids  = (const int*)  d_in[1];
  const int*   toff     = (const int*)  d_in[2];
  const float* emb      = (const float*)d_in[3];
  const float* WxF0 = (const float*)d_in[4];
  const float* WhF0 = (const float*)d_in[5];
  const float* bF0  = (const float*)d_in[6];
  const float* WxF1 = (const float*)d_in[7];
  const float* WhF1 = (const float*)d_in[8];
  const float* bF1  = (const float*)d_in[9];
  const float* WxB0 = (const float*)d_in[10];
  const float* WhB0 = (const float*)d_in[11];
  const float* bB0  = (const float*)d_in[12];
  const float* WxB1 = (const float*)d_in[13];
  const float* WhB1 = (const float*)d_in[14];
  const float* bB1  = (const float*)d_in[15];
  const float* aw   = (const float*)d_in[16];
  const float* ab   = (const float*)d_in[17];
  const float* ggk  = (const float*)d_in[18];
  const float* ggb  = (const float*)d_in[19];
  const float* gck  = (const float*)d_in[20];
  const float* gcb  = (const float*)d_in[21];
  const float* ow   = (const float*)d_in[22];
  const float* ob   = (const float*)d_in[23];

  char* ws = (char*)d_ws;
  size_t off = 0;
  auto alloc = [&](size_t bytes) { size_t r = off; off = (off + bytes + 255) & ~(size_t)255; return r; };

  __bf16* sent_bf = (__bf16*)(ws + alloc((size_t)NB*NT*NEP*2));
  __bf16* x1_bf   = (__bf16*)(ws + alloc((size_t)NB*NT*DIN2*2));
  float*  xwA     = (float*) (ws + alloc((size_t)NB*NT*N4H*4));
  float*  xwB     = (float*) (ws + alloc((size_t)NB*NT*N4H*4));
  float*  hf      = (float*) (ws + alloc((size_t)NB*NT*NH*4));
  float*  hb      = (float*) (ws + alloc((size_t)NB*NT*NH*4));
  float*  mem     = (float*) (ws + alloc((size_t)NB*NT*NDM*4));
  float*  vt      = (float*) (ws + alloc((size_t)NB*NE*4));
  float*  ep      = (float*) (ws + alloc((size_t)NB*NG*4));
  float*  sc      = (float*) (ws + alloc((size_t)NB*NT*4));
  float*  content = (float*) (ws + alloc((size_t)NB*NDM*4));
  float*  bc      = (float*) (ws + alloc((size_t)NB*4));
  __bf16* wxf0 = (__bf16*)(ws + alloc((size_t)NEP*N4H*2));
  __bf16* wxb0 = (__bf16*)(ws + alloc((size_t)NEP*N4H*2));
  __bf16* wxf1 = (__bf16*)(ws + alloc((size_t)DIN2*N4H*2));
  __bf16* wxb1 = (__bf16*)(ws + alloc((size_t)DIN2*N4H*2));
  __bf16* whf0 = (__bf16*)(ws + alloc((size_t)NH*N4H*2));
  __bf16* whb0 = (__bf16*)(ws + alloc((size_t)NH*N4H*2));
  __bf16* whf1 = (__bf16*)(ws + alloc((size_t)NH*N4H*2));
  __bf16* whb1 = (__bf16*)(ws + alloc((size_t)NH*N4H*2));

  const int lstm_lds = NB*N4H*4 + NB*NH*4 + NB*NH*2;  // 88 KB
  hipFuncSetAttribute((const void*)k_lstm,
                      hipFuncAttributeMaxDynamicSharedMemorySize, lstm_lds);

  // --- weight casts (fp32 -> bf16, K padded to multiple of 32) ---
  auto castw = [&](const float* s, __bf16* d, int K, int Kp, int N) {
    k_cast_w<<<(Kp*N + 255)/256, 256, 0, stream>>>(s, d, K, Kp, N);
  };
  castw(WxF0, wxf0, NE, NEP, N4H);  castw(WxB0, wxb0, NE, NEP, N4H);
  castw(WxF1, wxf1, DIN2, DIN2, N4H); castw(WxB1, wxb1, DIN2, DIN2, N4H);
  castw(WhF0, whf0, NH, NH, N4H);   castw(WhB0, whb0, NH, NH, N4H);
  castw(WhF1, whf1, NH, NH, N4H);   castw(WhB1, whb1, NH, NH, N4H);

  // --- embedding gather + target vector ---
  k_embed_sent<<<NB*NT, NEP, 0, stream>>>(sent_ids, emb, sent_bf);
  k_vtarget<<<NB, NEP, 0, stream>>>(tgt_ids, emb, vt);

  // --- layer 0: xW GEMMs + recurrence ---
  dim3 ggrid(N4H/128, (NB*NT)/128);
  k_gemm_xw<<<ggrid, 256, 0, stream>>>(sent_bf, wxf0, bF0, xwA, NEP);
  k_gemm_xw<<<ggrid, 256, 0, stream>>>(sent_bf, wxb0, bB0, xwB, NEP);
  k_lstm<<<2, 256, lstm_lds, stream>>>(xwA, xwB, whf0, whb0, hf, hb);

  // --- layer 1 ---
  k_concat_x1<<<NB*NT, 256, 0, stream>>>(hf, hb, x1_bf);
  k_gemm_xw<<<ggrid, 256, 0, stream>>>(x1_bf, wxf1, bF1, xwA, DIN2);
  k_gemm_xw<<<ggrid, 256, 0, stream>>>(x1_bf, wxb1, bB1, xwB, DIN2);
  k_lstm<<<2, 256, lstm_lds, stream>>>(xwA, xwB, whf1, whb1, hf, hb);

  // --- memory + episodic hops ---
  k_memory<<<NB*NT, 256, 0, stream>>>(hf, hb, toff, mem);
  k_zero<<<(NB*NG + 255)/256, 256, 0, stream>>>(ep, NB*NG);
  for (int hop = 0; hop < 3; ++hop) {
    k_bconst<<<NB, 128, 0, stream>>>(ep, vt, aw, ab, bc);
    k_scores<<<dim3(NT/256, NB), 256, 0, stream>>>(mem, aw, bc, sc);
    k_softmax_content<<<NB, 256, 0, stream>>>(mem, sc, content);
    k_hop<<<NB, 256, 0, stream>>>(content, ggk, ggb, gck, gcb, ep);
  }
  k_logits<<<1, 128, 0, stream>>>(ep, ow, ob, (float*)d_out);
}
